// GNN_11158325035416
// MI455X (gfx1250) — matrix operand order repaired and checked by reading, not compiled, not run on hardware
//
#include <hip/hip_runtime.h>
#include <hip/hip_bf16.h>

// GIN forward for MI455X (gfx1250): bf16 WMMA GEMMs with f32 accumulation,
// f32 hardware atomics for segment_sum, BN stats fused into GEMM epilogue.

typedef __bf16 bf16_t;
typedef __attribute__((ext_vector_type(16))) __bf16 v16bf;
typedef __attribute__((ext_vector_type(8)))  __bf16 v8bf;
typedef __attribute__((ext_vector_type(8)))  float  v8f;

static constexpr int kN = 131072;   // nodes
static constexpr int kE = 262144;   // edges
static constexpr int kD = 300;      // feature dim
static constexpr int kL = 5;        // layers
static constexpr int K1 = 320;      // 300 padded to 32-multiple (GEMM1 K)
static constexpr int C1 = 640;      // 600 padded (GEMM1 out cols / GEMM2 K)
static constexpr int C2 = 320;      // 300 padded (GEMM2 out cols)

__device__ __forceinline__ bf16_t f2bf(float f) {
  union { float f; unsigned u; } x; x.f = f;
  unsigned r = x.u + 0x7FFFu + ((x.u >> 16) & 1u);   // round-to-nearest-even
  unsigned short h = (unsigned short)(r >> 16);
  bf16_t out; __builtin_memcpy(&out, &h, 2); return out;
}
__device__ __forceinline__ float bf2f(bf16_t b) {
  unsigned short h; __builtin_memcpy(&h, &b, 2);
  unsigned u = ((unsigned)h) << 16;
  float f; __builtin_memcpy(&f, &u, 4); return f;
}

// h0[i,c] = x_emb1[x[i,0],c] + x_emb2[x[i,1],c], padded to stride 320
__global__ void embed_kernel(const int* __restrict__ x,
                             const float* __restrict__ xe1,
                             const float* __restrict__ xe2,
                             float* __restrict__ h) {
  int idx = blockIdx.x * blockDim.x + threadIdx.x;
  if (idx >= kN * C2) return;
  int i = idx / C2, c = idx - i * C2;
  float v = 0.f;
  if (c < kD) v = xe1[x[2 * i] * kD + c] + xe2[x[2 * i + 1] * kD + c];
  h[idx] = v;
}

__global__ void fill_kernel(float* __restrict__ p, int n) {
  int idx = blockIdx.x * blockDim.x + threadIdx.x;
  if (idx < n) p[idx] = 0.f;
}

// agg[dst] += h[src] + edge_emb1[l][ea0] + edge_emb2[l][ea1]
__global__ void scatter_kernel(const float* __restrict__ h,
                               const int* __restrict__ ei,
                               const int* __restrict__ ea,
                               const float* __restrict__ ee1,
                               const float* __restrict__ ee2,
                               float* __restrict__ agg) {
  int idx = blockIdx.x * blockDim.x + threadIdx.x;
  if (idx >= kE * kD) return;
  int e = idx / kD, c = idx - e * kD;
  int src = ei[e], dst = ei[kE + e];
  int a0 = ea[2 * e], a1 = ea[2 * e + 1];
  float v = h[(size_t)src * C2 + c] + ee1[a0 * kD + c] + ee2[a1 * kD + c];
  unsafeAtomicAdd(&agg[(size_t)dst * C2 + c], v);
}

// f32 [N,320] -> bf16 [N,320]
__global__ void convertA_kernel(const float* __restrict__ a, bf16_t* __restrict__ b) {
  int idx = blockIdx.x * blockDim.x + threadIdx.x;
  if (idx >= kN * K1) return;
  b[idx] = f2bf(a[idx]);
}

// W [Kv, Nv] row-major f32 -> Wt [NPAD, KPAD] bf16 (transposed + zero-padded)
__global__ void wtrans_kernel(const float* __restrict__ W, int Kv, int Nv, int ldw,
                              bf16_t* __restrict__ Wt, int KPAD, int total) {
  int idx = blockIdx.x * blockDim.x + threadIdx.x;
  if (idx >= total) return;
  int n = idx / KPAD, k = idx - n * KPAD;
  float v = (k < Kv && n < Nv) ? W[(size_t)k * ldw + n] : 0.f;
  Wt[idx] = f2bf(v);
}

// C[m, n] = sum_k A[m,k]*Bt[n,k] + bias[n]; also accumulates per-column
// sum / sumsq for the following BatchNorm. One wave -> 16 rows x 64 cols.
template <int KPAD, int LDC, bool OUT_BF16>
__global__ __launch_bounds__(256)
void gemm_kernel(const bf16_t* __restrict__ A, const bf16_t* __restrict__ Bt,
                 const float* __restrict__ bias, int nbias,
                 void* __restrict__ Cout,
                 float* __restrict__ ssum, float* __restrict__ ssq) {
  const int lane  = threadIdx.x & 31;
  const int wave  = threadIdx.x >> 5;
  const int m0    = (blockIdx.x * 8 + wave) * 16;
  const int n0    = blockIdx.y * 64;
  const int lh    = lane & 15;
  const int khalf = lane >> 4;   // 0: K low half, 1: K high half (contiguous k order, same for A and B)

  const bf16_t* arow = A + (size_t)(m0 + lh) * KPAD + khalf * 16;
  v8f acc[4] = {};

  for (int k = 0; k < KPAD; k += 32) {
    // A fragment: elems 0-7 = A[row, k + khalf*16 + 0..7]
    //             elems 8-15 = A[row, k + khalf*16 + 8 + 0..7]
    v8bf alo = *(const v8bf*)(arow + k);
    v8bf ahi = *(const v8bf*)(arow + k + 8);
    v16bf a;
#pragma unroll
    for (int e = 0; e < 8; ++e) { a[e] = alo[e]; a[8 + e] = ahi[e]; }
#pragma unroll
    for (int j = 0; j < 4; ++j) {
      const int col = n0 + j * 16 + lh;
      // B fragment: 16 contiguous K values of column `col` (Bt is [cols, KPAD])
      v16bf b = *(const v16bf*)(Bt + (size_t)col * KPAD + k + khalf * 16);
      acc[j] = __builtin_amdgcn_wmma_f32_16x16x32_bf16(
          false, a, false, b, (short)0, acc[j], false, false);
    }
  }

#pragma unroll
  for (int j = 0; j < 4; ++j) {
    const int col = n0 + j * 16 + lh;
    const float bi = (col < nbias) ? bias[col] : 0.f;
    float s = 0.f, q = 0.f;
#pragma unroll
    for (int i = 0; i < 8; ++i) {
      const int row = m0 + khalf * 8 + i;   // C layout: lane<16 rows 0-7, else 8-15
      float v = acc[j][i] + bi;
      s += v; q += v * v;
      if (OUT_BF16) ((bf16_t*)Cout)[(size_t)row * LDC + col] = f2bf(v);
      else          ((float*)Cout)[(size_t)row * LDC + col] = v;
    }
    unsafeAtomicAdd(&ssum[col], s);
    unsafeAtomicAdd(&ssq[col], q);
  }
}

// scale = g * rsqrt(var+eps); shift = b - mean*scale   (pads -> 0)
__global__ void finalize_kernel(const float* __restrict__ ssum, const float* __restrict__ ssq,
                                const float* __restrict__ g, const float* __restrict__ b,
                                int nvalid, float* __restrict__ scale,
                                float* __restrict__ shift, int ntot) {
  int c = blockIdx.x * blockDim.x + threadIdx.x;
  if (c >= ntot) return;
  if (c < nvalid) {
    float mean = ssum[c] * (1.f / kN);
    float var  = ssq[c] * (1.f / kN) - mean * mean;
    float inv  = rsqrtf(var + 1e-5f);
    float sc   = g[c] * inv;
    scale[c] = sc;
    shift[c] = b[c] - mean * sc;
  } else {
    scale[c] = 0.f; shift[c] = 0.f;
  }
}

// In-place BN + ReLU on bf16 [N, C1] (pads normalize to 0 since scale/shift=0)
__global__ void norm_bf16_kernel(bf16_t* __restrict__ z, const float* __restrict__ scale,
                                 const float* __restrict__ shift, int total) {
  int idx = blockIdx.x * blockDim.x + threadIdx.x;
  if (idx >= total) return;
  int c = idx & (C1 - 1);   // C1 = 640? not pow2 -> use mod
  c = idx - (idx / C1) * C1;
  float y = fmaxf(bf2f(z[idx]) * scale[c] + shift[c], 0.f);
  z[idx] = f2bf(y);
}

// BN (+optional ReLU) of f32 z [N,320] -> out [N,ldout] f32
__global__ void norm_out_kernel(const float* __restrict__ z, const float* __restrict__ scale,
                                const float* __restrict__ shift, float* __restrict__ out,
                                int ldout, int relu, int total) {
  int idx = blockIdx.x * blockDim.x + threadIdx.x;
  if (idx >= total) return;
  int i = idx / ldout, c = idx - i * ldout;
  float v = 0.f;
  if (c < kD) {
    v = z[(size_t)i * C2 + c] * scale[c] + shift[c];
    if (relu) v = fmaxf(v, 0.f);
  }
  out[idx] = v;
}

extern "C" void kernel_launch(void* const* d_in, const int* in_sizes, int n_in,
                              void* d_out, int out_size, void* d_ws, size_t ws_size,
                              hipStream_t stream) {
  const int*   x   = (const int*)d_in[0];
  const int*   ei  = (const int*)d_in[1];
  const int*   ea  = (const int*)d_in[2];
  const float* xe1 = (const float*)d_in[3];
  const float* xe2 = (const float*)d_in[4];
  const float* ee1 = (const float*)d_in[5];   // [L,5,300]
  const float* ee2 = (const float*)d_in[6];   // [L,3,300]
  const float* W1  = (const float*)d_in[7];   // [L,300,600]
  const float* b1  = (const float*)d_in[8];   // [L,600]
  const float* g1  = (const float*)d_in[9];
  const float* bb1 = (const float*)d_in[10];
  const float* W2  = (const float*)d_in[11];  // [L,600,300]
  const float* b2  = (const float*)d_in[12];
  const float* g2  = (const float*)d_in[13];
  const float* bb2 = (const float*)d_in[14];
  float* out = (float*)d_out;

  // workspace layout (~590 MB)
  char* ws = (char*)d_ws;
  float*  hbuf = (float*)ws;                                  // [N,320] f32
  float*  aggF = hbuf + (size_t)kN * C2;                      // [N,320] f32 (reused as z2)
  bf16_t* aggB = (bf16_t*)(aggF + (size_t)kN * C2);           // [N,320] bf16
  bf16_t* z1B  = aggB + (size_t)kN * K1;                      // [N,640] bf16
  bf16_t* W1t  = z1B + (size_t)kN * C1;                       // [640,320] bf16
  bf16_t* W2t  = W1t + (size_t)C1 * K1;                       // [320,640] bf16
  float*  stats = (float*)(W2t + (size_t)C2 * C1);
  float *ssum = stats, *ssq = stats + 640, *scale = stats + 1280, *shift = stats + 1920;

  const int TB = 256;
  auto cdiv = [](long a, long b) { return (int)((a + b - 1) / b); };

  embed_kernel<<<cdiv((long)kN * C2, TB), TB, 0, stream>>>(x, xe1, xe2, hbuf);

  for (int l = 0; l < kL; ++l) {
    // --- aggregate: agg = segment_sum(h[src] + ee, dst) ---
    fill_kernel<<<cdiv((long)kN * C2, TB), TB, 0, stream>>>(aggF, kN * C2);
    scatter_kernel<<<cdiv((long)kE * kD, TB), TB, 0, stream>>>(
        hbuf, ei, ea, ee1 + (size_t)l * 5 * kD, ee2 + (size_t)l * 3 * kD, aggF);
    convertA_kernel<<<cdiv((long)kN * K1, TB), TB, 0, stream>>>(aggF, aggB);

    // --- GEMM1 + bias + stats ---
    wtrans_kernel<<<cdiv((long)C1 * K1, TB), TB, 0, stream>>>(
        W1 + (size_t)l * kD * 600, kD, 600, 600, W1t, K1, C1 * K1);
    fill_kernel<<<cdiv(1280L, TB), TB, 0, stream>>>(ssum, 1280);
    dim3 grid1(kN / 128, C1 / 64);
    gemm_kernel<K1, C1, true><<<grid1, 256, 0, stream>>>(
        aggB, W1t, b1 + (size_t)l * 600, 600, z1B, ssum, ssq);
    finalize_kernel<<<cdiv(640L, TB), TB, 0, stream>>>(
        ssum, ssq, g1 + (size_t)l * 600, bb1 + (size_t)l * 600, 600, scale, shift, 640);
    norm_bf16_kernel<<<cdiv((long)kN * C1, TB), TB, 0, stream>>>(z1B, scale, shift, kN * C1);

    // --- GEMM2 + bias + stats ---
    wtrans_kernel<<<cdiv((long)C2 * C1, TB), TB, 0, stream>>>(
        W2 + (size_t)l * 600 * kD, 600, kD, kD, W2t, C1, C2 * C1);
    fill_kernel<<<cdiv(1280L, TB), TB, 0, stream>>>(ssum, 1280);
    dim3 grid2(kN / 128, C2 / 64);
    gemm_kernel<C1, C2, false><<<grid2, 256, 0, stream>>>(
        z1B, W2t, b2 + (size_t)l * kD, kD, aggF, ssum, ssq);
    finalize_kernel<<<cdiv(640L, TB), TB, 0, stream>>>(
        ssum, ssq, g2 + (size_t)l * kD, bb2 + (size_t)l * kD, kD, scale, shift, C2);

    // --- outer BN (+ReLU for inner layers) ---
    if (l < kL - 1) {
      norm_out_kernel<<<cdiv((long)kN * C2, TB), TB, 0, stream>>>(
          aggF, scale, shift, hbuf, C2, 1, kN * C2);
    } else {
      norm_out_kernel<<<cdiv((long)kN * kD, TB), TB, 0, stream>>>(
          aggF, scale, shift, out, kD, 0, kN * kD);
    }
  }
}
